// SinglePLEMDLayer_86955907875085
// MI455X (gfx1250) — matrix operand order repaired and checked by reading, not compiled, not run on hardware
//
#include <hip/hip_runtime.h>
#include <math.h>

// ---------------- problem constants ----------------
#define BQ   16384
#define DIN  512
#define NH1  256
#define NH2  128
#define NS   4
#define NDOM 8
#define NT   3
#define NTE  2
#define NEXP 18          // 4 shared + 8 domain + 6 task
#define GH   64
#define ETOT 14

typedef __bf16 v16bf __attribute__((ext_vector_type(16)));
typedef __bf16 v8bf  __attribute__((ext_vector_type(8)));
typedef float  v8f   __attribute__((ext_vector_type(8)));

// ---------------- workspace layout (bytes, 256-aligned) ----------------
#define OFF_XB    ((size_t)0)                         // B*DIN bf16           = 16,777,216
#define OFF_W1T   ((size_t)16777216)                  // 18*256*512 bf16      =  4,718,592
#define OFF_W2T   ((size_t)21495808)                  // 18*128*256 bf16      =  1,179,648
#define OFF_GW1T  ((size_t)22675456)                  // 3*64*512 bf16        =    196,608
#define OFF_EOUT  ((size_t)22872064)                  // 18*B*128 bf16        = 75,497,472
#define OFF_GATES ((size_t)98369536)                  // B*3*14 f32           =  2,752,512
// total ~101.2 MB

// ---------------- helpers ----------------
__device__ inline v16bf load_afrag(const __bf16* p) {
  // A-matrix 16x32 bf16 per-lane fragment: 16 contiguous K at K0
  v8bf lo = *(const v8bf*)p;
  v8bf hi = *(const v8bf*)(p + 8);
  return __builtin_shufflevector(lo, hi, 0,1,2,3,4,5,6,7,8,9,10,11,12,13,14,15);
}

__device__ inline v8f vzero8() {
  v8f z = {0.f,0.f,0.f,0.f,0.f,0.f,0.f,0.f};
  return z;
}

// ---------------- prep: x f32 -> bf16 ----------------
__global__ void prep_x_kernel(const float* __restrict__ x, __bf16* __restrict__ xb, int n) {
  int i = blockIdx.x * blockDim.x + threadIdx.x;
  if (i < n) xb[i] = (__bf16)x[i];
}

// ---------------- prep: transpose weights to [E][N][K] bf16 ----------------
__global__ void prep_w_kernel(const float* __restrict__ sW1, const float* __restrict__ dW1,
                              const float* __restrict__ tW1,
                              const float* __restrict__ sW2, const float* __restrict__ dW2,
                              const float* __restrict__ tW2,
                              const float* __restrict__ gW1,
                              __bf16* __restrict__ w1t, __bf16* __restrict__ w2t,
                              __bf16* __restrict__ gw1t) {
  const int N1 = NEXP * NH1 * DIN;   // 2,359,296
  const int N2 = NEXP * NH2 * NH1;   //   589,824
  const int N3 = NT * GH * DIN;      //    98,304
  int i = blockIdx.x * blockDim.x + threadIdx.x;
  if (i < N1) {
    int e = i / (NH1 * DIN); int r = i % (NH1 * DIN);
    int n = r / DIN;         int k = r % DIN;
    const float* w = (e < NS) ? (sW1 + (size_t)e * DIN * NH1)
                  : (e < NS + NDOM) ? (dW1 + (size_t)(e - NS) * DIN * NH1)
                  : (tW1 + (size_t)(e - NS - NDOM) * DIN * NH1);
    w1t[i] = (__bf16)w[(size_t)k * NH1 + n];
  } else if (i < N1 + N2) {
    int j = i - N1;
    int e = j / (NH2 * NH1); int r = j % (NH2 * NH1);
    int n = r / NH1;         int k = r % NH1;
    const float* w = (e < NS) ? (sW2 + (size_t)e * NH1 * NH2)
                  : (e < NS + NDOM) ? (dW2 + (size_t)(e - NS) * NH1 * NH2)
                  : (tW2 + (size_t)(e - NS - NDOM) * NH1 * NH2);
    w2t[j] = (__bf16)w[(size_t)k * NH2 + n];
  } else if (i < N1 + N2 + N3) {
    int j = i - N1 - N2;
    int t = j / (GH * DIN); int r = j % (GH * DIN);
    int n = r / DIN;        int k = r % DIN;
    gw1t[j] = (__bf16)gW1[(size_t)t * DIN * GH + (size_t)k * GH + n];
  }
}

// ---------------- fused 2-layer expert MLP via WMMA ----------------
#define L1STRIDE (NH1 + 8)   // 264 bf16 -> 528B row stride, 16B-aligned

__global__ __launch_bounds__(256)
void expert_kernel(const __bf16* __restrict__ xb,
                   const __bf16* __restrict__ w1t,
                   const __bf16* __restrict__ w2t,
                   const float* __restrict__ sb1, const float* __restrict__ db1,
                   const float* __restrict__ tb1,
                   const float* __restrict__ sb2, const float* __restrict__ db2,
                   const float* __restrict__ tb2,
                   __bf16* __restrict__ eout) {
  __shared__ __bf16 ldsH[128 * L1STRIDE];   // 67,584 bytes

  const int wave = threadIdx.x >> 5;
  const int lane = threadIdx.x & 31;
  const int e = blockIdx.y;
  const int rowBase = blockIdx.x * 128;

  const int M   = lane & 15;        // A row within tile
  const int hi  = lane >> 4;
  const int K0a = hi * 16;          // A K base
  const int Nb  = lane & 15;        // B col within tile
  const int K0b = hi * 16;          // B K base
  const int mB  = hi * 8;           // C/D row base for this lane

  const float* b1 = (e < NS) ? (sb1 + e * NH1)
                  : (e < NS + NDOM) ? (db1 + (e - NS) * NH1)
                  : (tb1 + (e - NS - NDOM) * NH1);
  const float* b2 = (e < NS) ? (sb2 + e * NH2)
                  : (e < NS + NDOM) ? (db2 + (e - NS) * NH2)
                  : (tb2 + (e - NS - NDOM) * NH2);

  const __bf16* arow  = xb  + (size_t)(rowBase + wave * 16 + M) * DIN + K0a;
  const __bf16* bcol  = w1t + ((size_t)e * NH1 + Nb) * DIN + K0b;

  // ---------- layer 1: H[128x256] = relu(X[128x512] @ W1[512x256] + b1) ----------
  // 16 col-tiles per wave, processed as 4 groups of 4 with double-buffered B frags.
  #pragma unroll
  for (int g = 0; g < 4; ++g) {
    v8f acc[4];
    #pragma unroll
    for (int i = 0; i < 4; ++i) acc[i] = vzero8();

    const __bf16* bp[4];
    v16bf bcur[4];
    #pragma unroll
    for (int ct = 0; ct < 4; ++ct) {
      bp[ct] = bcol + (size_t)((g * 4 + ct) * 16) * DIN;
      bcur[ct] = *(const v16bf*)bp[ct];
    }

    #pragma unroll
    for (int kb = 0; kb < 16; ++kb) {
      v16bf afrag = load_afrag(arow + kb * 32);
      v16bf bnext[4];
      if (kb < 15) {
        #pragma unroll
        for (int ct = 0; ct < 4; ++ct)
          bnext[ct] = *(const v16bf*)(bp[ct] + (kb + 1) * 32);
      }
      #pragma unroll
      for (int ct = 0; ct < 4; ++ct)
        acc[ct] = __builtin_amdgcn_wmma_f32_16x16x32_bf16(
            false, afrag, false, bcur[ct], (short)0, acc[ct], false, false);
      if (kb < 15) {
        #pragma unroll
        for (int ct = 0; ct < 4; ++ct) bcur[ct] = bnext[ct];
      }
    }

    #pragma unroll
    for (int ct = 0; ct < 4; ++ct) {
      int col = (g * 4 + ct) * 16 + Nb;
      float bias = b1[col];
      #pragma unroll
      for (int v = 0; v < 8; ++v) {
        float h = acc[ct][v] + bias;
        h = h > 0.f ? h : 0.f;
        ldsH[(wave * 16 + mB + v) * L1STRIDE + col] = (__bf16)h;
      }
    }
  }
  __syncthreads();

  // ---------- layer 2: O[128x128] = relu(H[128x256] @ W2[256x128] + b2) ----------
  const __bf16* hrow   = ldsH + (wave * 16 + M) * L1STRIDE + K0a;
  const __bf16* b2col  = w2t + ((size_t)e * NH2 + Nb) * NH1 + K0b;

  #pragma unroll
  for (int g = 0; g < 2; ++g) {
    v8f acc2[4];
    #pragma unroll
    for (int i = 0; i < 4; ++i) acc2[i] = vzero8();

    const __bf16* bp[4];
    v16bf bcur[4];
    #pragma unroll
    for (int ct = 0; ct < 4; ++ct) {
      bp[ct] = b2col + (size_t)((g * 4 + ct) * 16) * NH1;
      bcur[ct] = *(const v16bf*)bp[ct];
    }

    #pragma unroll
    for (int kb = 0; kb < 8; ++kb) {
      v16bf afrag = load_afrag(hrow + kb * 32);
      v16bf bnext[4];
      if (kb < 7) {
        #pragma unroll
        for (int ct = 0; ct < 4; ++ct)
          bnext[ct] = *(const v16bf*)(bp[ct] + (kb + 1) * 32);
      }
      #pragma unroll
      for (int ct = 0; ct < 4; ++ct)
        acc2[ct] = __builtin_amdgcn_wmma_f32_16x16x32_bf16(
            false, afrag, false, bcur[ct], (short)0, acc2[ct], false, false);
      if (kb < 7) {
        #pragma unroll
        for (int ct = 0; ct < 4; ++ct) bcur[ct] = bnext[ct];
      }
    }

    #pragma unroll
    for (int ct = 0; ct < 4; ++ct) {
      int col = (g * 4 + ct) * 16 + Nb;
      float bias = b2[col];
      #pragma unroll
      for (int v = 0; v < 8; ++v) {
        float o = acc2[ct][v] + bias;
        o = o > 0.f ? o : 0.f;
        eout[((size_t)e * BQ + rowBase + wave * 16 + mB + v) * NH2 + col] = (__bf16)o;
      }
    }
  }
}

// ---------------- gates: WMMA 512->64 + ReLU, then 64->14 + softmax ----------------
#define GSTRIDE (GH + 1)  // 65 floats

__global__ __launch_bounds__(256)
void gate_kernel(const __bf16* __restrict__ xb, const __bf16* __restrict__ gw1t,
                 const float* __restrict__ gb1, const float* __restrict__ gW2,
                 const float* __restrict__ gb2, float* __restrict__ gates) {
  __shared__ float gh[128 * GSTRIDE];  // 33,280 bytes

  const int wave = threadIdx.x >> 5;
  const int lane = threadIdx.x & 31;
  const int t = blockIdx.y;
  const int rowBase = blockIdx.x * 128;

  const int M   = lane & 15;
  const int hiL = lane >> 4;
  const int K0a = hiL * 16;
  const int Nb  = lane & 15;
  const int K0b = hiL * 16;
  const int mB  = hiL * 8;

  v8f acc[4];
  #pragma unroll
  for (int i = 0; i < 4; ++i) acc[i] = vzero8();

  const __bf16* arow  = xb + (size_t)(rowBase + wave * 16 + M) * DIN + K0a;
  const __bf16* bcol  = gw1t + ((size_t)t * GH + Nb) * DIN + K0b;

  const __bf16* bp[4];
  v16bf bcur[4];
  #pragma unroll
  for (int ct = 0; ct < 4; ++ct) {
    bp[ct] = bcol + (size_t)(ct * 16) * DIN;
    bcur[ct] = *(const v16bf*)bp[ct];
  }

  #pragma unroll
  for (int kb = 0; kb < 16; ++kb) {
    v16bf afrag = load_afrag(arow + kb * 32);
    v16bf bnext[4];
    if (kb < 15) {
      #pragma unroll
      for (int ct = 0; ct < 4; ++ct)
        bnext[ct] = *(const v16bf*)(bp[ct] + (kb + 1) * 32);
    }
    #pragma unroll
    for (int ct = 0; ct < 4; ++ct)
      acc[ct] = __builtin_amdgcn_wmma_f32_16x16x32_bf16(
          false, afrag, false, bcur[ct], (short)0, acc[ct], false, false);
    if (kb < 15) {
      #pragma unroll
      for (int ct = 0; ct < 4; ++ct) bcur[ct] = bnext[ct];
    }
  }

  #pragma unroll
  for (int ct = 0; ct < 4; ++ct) {
    int col = ct * 16 + Nb;
    float bias = gb1[t * GH + col];
    #pragma unroll
    for (int v = 0; v < 8; ++v) {
      float h = acc[ct][v] + bias;
      gh[(wave * 16 + mB + v) * GSTRIDE + col] = h > 0.f ? h : 0.f;
    }
  }
  __syncthreads();

  if (threadIdx.x < 128) {
    int row = threadIdx.x;
    float logit[ETOT];
    float mx = -3.0e38f;
    #pragma unroll
    for (int et = 0; et < ETOT; ++et) {
      float s = gb2[t * ETOT + et];
      for (int g = 0; g < GH; ++g)
        s += gh[row * GSTRIDE + g] * gW2[((size_t)t * GH + g) * ETOT + et];
      logit[et] = s;
      mx = fmaxf(mx, s);
    }
    float sum = 0.f;
    #pragma unroll
    for (int et = 0; et < ETOT; ++et) {
      logit[et] = __expf(logit[et] - mx);
      sum += logit[et];
    }
    float inv = 1.f / sum;
    float* g = gates + ((size_t)(rowBase + row) * NT + t) * ETOT;
    #pragma unroll
    for (int et = 0; et < ETOT; ++et) g[et] = logit[et] * inv;
  }
}

// ---------------- gate-weighted mixing ----------------
__global__ void mix_kernel(const __bf16* __restrict__ eout,
                           const float* __restrict__ gates,
                           float* __restrict__ out) {
  int idx = blockIdx.x * blockDim.x + threadIdx.x;   // over B*H2
  if (idx >= BQ * NH2) return;
  int b = idx >> 7;
  int h = idx & (NH2 - 1);

  float base[NS + NDOM];
  #pragma unroll
  for (int e = 0; e < NS + NDOM; ++e)
    base[e] = (float)eout[((size_t)e * BQ + b) * NH2 + h];

  #pragma unroll
  for (int t = 0; t < NT; ++t) {
    const float* g = gates + ((size_t)b * NT + t) * ETOT;
    float s = 0.f;
    #pragma unroll
    for (int e = 0; e < NS + NDOM; ++e) s += g[e] * base[e];
    int te0 = NS + NDOM + t * NTE;           // global task-expert index
    s += g[12] * (float)eout[((size_t)te0 * BQ + b) * NH2 + h];
    s += g[13] * (float)eout[((size_t)(te0 + 1) * BQ + b) * NH2 + h];
    out[(size_t)t * BQ * NH2 + idx] = s;
  }
}

// ---------------- launch ----------------
extern "C" void kernel_launch(void* const* d_in, const int* in_sizes, int n_in,
                              void* d_out, int out_size, void* d_ws, size_t ws_size,
                              hipStream_t stream) {
  const float* x   = (const float*)d_in[0];
  // d_in[1] = domain_ids (unused by outputs)
  const float* sW1 = (const float*)d_in[2];
  const float* sb1 = (const float*)d_in[3];
  const float* sW2 = (const float*)d_in[4];
  const float* sb2 = (const float*)d_in[5];
  const float* dW1 = (const float*)d_in[6];
  const float* db1 = (const float*)d_in[7];
  const float* dW2 = (const float*)d_in[8];
  const float* db2 = (const float*)d_in[9];
  const float* tW1 = (const float*)d_in[10];
  const float* tb1 = (const float*)d_in[11];
  const float* tW2 = (const float*)d_in[12];
  const float* tb2 = (const float*)d_in[13];
  const float* gW1 = (const float*)d_in[14];
  const float* gb1 = (const float*)d_in[15];
  const float* gW2 = (const float*)d_in[16];
  const float* gb2 = (const float*)d_in[17];

  char* ws = (char*)d_ws;
  __bf16* xb    = (__bf16*)(ws + OFF_XB);
  __bf16* w1t   = (__bf16*)(ws + OFF_W1T);
  __bf16* w2t   = (__bf16*)(ws + OFF_W2T);
  __bf16* gw1t  = (__bf16*)(ws + OFF_GW1T);
  __bf16* eoutp = (__bf16*)(ws + OFF_EOUT);
  float*  gates = (float*)(ws + OFF_GATES);
  float*  out   = (float*)d_out;

  // prep
  {
    int n = BQ * DIN;
    prep_x_kernel<<<(n + 255) / 256, 256, 0, stream>>>(x, xb, n);
    int nw = NEXP * NH1 * DIN + NEXP * NH2 * NH1 + NT * GH * DIN;
    prep_w_kernel<<<(nw + 255) / 256, 256, 0, stream>>>(
        sW1, dW1, tW1, sW2, dW2, tW2, gW1, w1t, w2t, gw1t);
  }

  // experts: 128 row-tiles x 18 experts
  expert_kernel<<<dim3(BQ / 128, NEXP), 256, 0, stream>>>(
      xb, w1t, w2t, sb1, db1, tb1, sb2, db2, tb2, eoutp);

  // gates: 128 row-tiles x 3 tasks
  gate_kernel<<<dim3(BQ / 128, NT), 256, 0, stream>>>(
      xb, gw1t, gb1, gW2, gb2, gates);

  // mixing
  {
    int n = BQ * NH2;
    mix_kernel<<<(n + 255) / 256, 256, 0, stream>>>(eoutp, gates, out);
  }
}
